// multihead_attention_12524124635982
// MI455X (gfx1250) — matrix operand order repaired and checked by reading, not compiled, not run on hardware
//
#include <hip/hip_runtime.h>

typedef _Float16 f16;
typedef __attribute__((ext_vector_type(16))) _Float16 v16h;
typedef __attribute__((ext_vector_type(8)))  _Float16 v8h;
typedef __attribute__((ext_vector_type(8)))  float    v8f;
typedef __attribute__((ext_vector_type(4)))  float    v4f;

#define NHEADS 8
#define BB 4
#define TT 2048
#define DD 512
#define DH 64
#define NEG_INF (-4294967295.0f)

// Async copy of 16 bytes global -> LDS (CDNA5, tracked by ASYNCcnt).
__device__ __forceinline__ void async_g2l_b128(const void* gptr, void* lptr) {
    uint32_t ldso = (uint32_t)(uintptr_t)lptr;          // LDS byte offset
    uint64_t ga   = (uint64_t)(uintptr_t)gptr;
    asm volatile("global_load_async_to_lds_b128 %0, %1, off"
                 :: "v"(ldso), "v"(ga) : "memory");
}
__device__ __forceinline__ void wait_async0() {
    asm volatile("s_wait_asynccnt 0" ::: "memory");
}

// xor-permute within 16-lane groups on the VALU (v_permlane16_b32),
// keeping the row reductions off the DS pipe.
template <int M>
__device__ __forceinline__ float lanexor16(float x) {
#if __has_builtin(__builtin_amdgcn_permlane16)
    constexpr unsigned S1 = (M == 1) ? 0x67452301u : (M == 2) ? 0x54761032u
                          : (M == 4) ? 0x32107654u : 0xFEDCBA98u;
    constexpr unsigned S2 = (M == 1) ? 0xEFCDAB89u : (M == 2) ? 0xDCFE98BAu
                          : (M == 4) ? 0xBA98FEDCu : 0x76543210u;
    int i = __builtin_bit_cast(int, x);
    i = __builtin_amdgcn_permlane16(i, i, (int)S1, (int)S2, false, false);
    return __builtin_bit_cast(float, i);
#else
    return __shfl_xor(x, M, 32);
#endif
}

// ---------------------------------------------------------------------------
// Kernel 1: key/query padding masks: mask[b][t] = (sum_d x[b][t][d] != 0)
// ---------------------------------------------------------------------------
__global__ __launch_bounds__(256)
void mask_kernel(const float* __restrict__ q, const float* __restrict__ k,
                 float* __restrict__ qmask, float* __restrict__ kmask) {
    int row  = blockIdx.x * 8 + (threadIdx.x >> 5);   // 0..16383
    int lane = threadIdx.x & 31;
    const float* src = (row < BB * TT) ? q : k;
    float*       dst = (row < BB * TT) ? qmask : kmask;
    int r = row & (BB * TT - 1);
    const float* p = src + (size_t)r * DD;
    float s = 0.f;
    for (int i = lane; i < DD; i += 32) s += p[i];
#pragma unroll
    for (int m = 16; m >= 1; m >>= 1) s += __shfl_xor(s, m, 32);
    if (lane == 0) dst[r] = (s != 0.0f) ? 1.0f : 0.0f;
}

// ---------------------------------------------------------------------------
// Kernel 2: fused GEMM + bias + ReLU (+uniform output scale), f32 -> f16.
// LAYOUT 0: out[(head*B+b)][t][d] (Q,K).  LAYOUT 1: out[(head*B+b)][d][t] (V).
// oscale folds the attention 1/sqrt(dh)=0.125 into Q exactly (power of two).
// ---------------------------------------------------------------------------
template <int LAYOUT>
__global__ __launch_bounds__(128)
void proj_kernel(const float* __restrict__ X, const float* __restrict__ W,
                 const float* __restrict__ bias, f16* __restrict__ out,
                 float oscale) {
    __shared__ f16 Xs[64][40];   // [m_local][k_local]  (k chunk of 32)
    __shared__ f16 Ws[64][40];   // W^T tile: [n_local][k_local]

    const int tid  = threadIdx.x;
    const int wave = tid >> 5;
    const int lane = tid & 31;
    const int g2   = lane >> 4;
    const int lm   = lane & 15;
    const int n0   = blockIdx.x * 64;
    const int m0   = blockIdx.y * 64;

    v8f acc[4];
#pragma unroll
    for (int j = 0; j < 4; ++j)
#pragma unroll
        for (int v = 0; v < 8; ++v) acc[j][v] = 0.f;

    for (int k0 = 0; k0 < DD; k0 += 32) {
        __syncthreads();
        {   // stage X tile (64x32 f32 -> f16)
            const int r  = tid >> 1;
            const int c0 = (tid & 1) * 16;
            const float* g = X + (size_t)(m0 + r) * DD + k0 + c0;
#pragma unroll
            for (int i = 0; i < 16; i += 4) {
                v4f v = *(const v4f*)(g + i);
#pragma unroll
                for (int u = 0; u < 4; ++u) Xs[r][c0 + i + u] = (f16)v[u];
            }
        }
        {   // stage W^T tile: Ws[n][kk] = W[k0+kk][n0+n]
            const int n   = tid >> 1;
            const int kk0 = (tid & 1) * 16;
            const float* g = W + (size_t)(k0 + kk0) * DD + n0 + n;
#pragma unroll
            for (int i = 0; i < 16; ++i) Ws[n][kk0 + i] = (f16)g[(size_t)i * DD];
        }
        __syncthreads();

        v16h a;
        {
            const f16* p0 = &Xs[wave * 16 + lm][16 * g2];
            const f16* p1 = &Xs[wave * 16 + lm][16 * g2 + 8];
            v8h lo = *(const v8h*)p0;
            v8h hi = *(const v8h*)p1;
#pragma unroll
            for (int i = 0; i < 8; ++i) { a[i] = lo[i]; a[i + 8] = hi[i]; }
        }
#pragma unroll
        for (int j = 0; j < 4; ++j) {
            const f16* p = &Ws[j * 16 + lm][16 * g2];
            v8h lo = *(const v8h*)p;
            v8h hi = *(const v8h*)(p + 8);
            v16h bf;
#pragma unroll
            for (int i = 0; i < 8; ++i) { bf[i] = lo[i]; bf[i + 8] = hi[i]; }
            acc[j] = __builtin_amdgcn_wmma_f32_16x16x32_f16(
                false, a, false, bf, (short)0, acc[j], false, false);
        }
    }

#pragma unroll
    for (int j = 0; j < 4; ++j) {
        const int n    = n0 + j * 16 + lm;
        const float bv = bias[n];
        const int head = n >> 6;
        const int d    = n & 63;
#pragma unroll
        for (int v = 0; v < 8; ++v) {
            const int r = m0 + wave * 16 + v + 8 * g2;
            const int b = r >> 11;
            const int t = r & (TT - 1);
            float val = fmaxf(acc[j][v] + bv, 0.f) * oscale;
            size_t idx = (LAYOUT == 0)
                ? ((size_t)((head * BB + b) * TT + t) * DH + d)
                : ((size_t)((head * BB + b) * DH + d) * TT + t);
            out[idx] = (f16)val;
        }
    }
}

// ---------------------------------------------------------------------------
// Kernel 3: flash-attention.  4 waves/block share a head-batch; K/V tiles are
// double-buffered in LDS via global_load_async_to_lds_b128 so the copy of
// tile kb+32 overlaps compute on tile kb.  Row sums go through an extra WMMA
// against a ones matrix (lands directly in the per-lane stat layout); row max
// uses v_permlane16 xor steps on the VALU.
// ---------------------------------------------------------------------------
__device__ __forceinline__ void stage_tile(const f16* __restrict__ K,
                                           const f16* __restrict__ V, int kb,
                                           f16 (*Ksb)[72], f16 (*Vsb)[40],
                                           int tid) {
#pragma unroll
    for (int c = tid; c < 256; c += 128) {        // K: 32x64 f16, 16B chunks
        const int key = c >> 3;
        const int off = (c & 7) * 8;
        async_g2l_b128(K + (size_t)(kb + key) * DH + off, &Ksb[key][off]);
    }
#pragma unroll
    for (int c = tid; c < 256; c += 128) {        // V^T: 64x32 f16, 16B chunks
        const int d   = c >> 2;
        const int off = (c & 3) * 8;
        async_g2l_b128(V + (size_t)d * TT + kb + off, &Vsb[d][off]);
    }
}

__global__ __launch_bounds__(128)
void attn_kernel(const f16* __restrict__ Qh, const f16* __restrict__ Kh,
                 const f16* __restrict__ Vt,
                 const float* __restrict__ kmask, const float* __restrict__ qmask,
                 float* __restrict__ attnOut) {
    __shared__ f16 Ks[2][32][72];     // double-buffered key tiles (144B rows)
    __shared__ f16 Vs[2][64][40];     // double-buffered V^T tiles (80B rows)
    __shared__ f16 Plds[4][16][40];   // wave-private P staging

    const int tid  = threadIdx.x;
    const int wave = tid >> 5;
    const int lane = tid & 31;
    const int g2   = lane >> 4;
    const int lm   = lane & 15;
    const int hb   = blockIdx.x >> 5;               // head*B + b  (0..31)
    const int q0   = ((blockIdx.x & 31) * 4 + wave) * 16;
    const int b    = hb & (BB - 1);

    const f16* Q = Qh + (size_t)hb * TT * DH;
    const f16* K = Kh + (size_t)hb * TT * DH;
    const f16* V = Vt + (size_t)hb * DH * TT;
    const float* km = kmask + (size_t)b * TT;

    // Q A-fragments (pre-scaled by 0.125 at projection time), resident
    v16h aQ[2];
    {
        const f16* qp = Q + (size_t)(q0 + lm) * DH;
#pragma unroll
        for (int c = 0; c < 2; ++c) {
            v8h lo = *(const v8h*)(qp + c * 32 + 16 * g2);
            v8h hi = *(const v8h*)(qp + c * 32 + 16 * g2 + 8);
#pragma unroll
            for (int i = 0; i < 8; ++i) { aQ[c][i] = lo[i]; aQ[c][i + 8] = hi[i]; }
        }
    }

    v16h ones;     // B matrix of ones for the row-sum WMMA
#pragma unroll
    for (int i = 0; i < 16; ++i) ones[i] = (f16)1.0f;
    v8f zero8;
#pragma unroll
    for (int v = 0; v < 8; ++v) zero8[v] = 0.f;

    v8f accO[4];
#pragma unroll
    for (int j = 0; j < 4; ++j)
#pragma unroll
        for (int v = 0; v < 8; ++v) accO[j][v] = 0.f;
    float mrow[8], lrow[8];
#pragma unroll
    for (int v = 0; v < 8; ++v) { mrow[v] = -INFINITY; lrow[v] = 0.f; }

    // prologue: load tile 0
    stage_tile(K, V, 0, Ks[0], Vs[0], tid);
    wait_async0();
    __syncthreads();

    int cur = 0;
    for (int kb = 0; kb < TT; kb += 32) {
        // prefetch next tile into the other buffer (overlaps compute below)
        if (kb + 32 < TT) stage_tile(K, V, kb + 32, Ks[cur ^ 1], Vs[cur ^ 1], tid);

        // ---- scores S = (Q/8).K^T, two 16-key column groups ----
        v8f s[2];
#pragma unroll
        for (int jn = 0; jn < 2; ++jn) {
            const f16* kp = &Ks[cur][jn * 16 + lm][16 * g2];
            v8f c = zero8;
#pragma unroll
            for (int kc = 0; kc < 2; ++kc) {
                v8h lo = *(const v8h*)(kp + kc * 32);
                v8h hi = *(const v8h*)(kp + kc * 32 + 8);
                v16h bf;
#pragma unroll
                for (int i = 0; i < 8; ++i) { bf[i] = lo[i]; bf[i + 8] = hi[i]; }
                c = __builtin_amdgcn_wmma_f32_16x16x32_f16(
                    false, aQ[kc], false, bf, (short)0, c, false, false);
            }
            const float kmv = km[kb + jn * 16 + lm];
#pragma unroll
            for (int v = 0; v < 8; ++v)
                s[jn][v] = (kmv == 0.f) ? NEG_INF : c[v];
        }

        // ---- row max via VALU permlane xor steps (16-lane C halves) ----
        float pscale[8];
#pragma unroll
        for (int v = 0; v < 8; ++v) {
            float t = fmaxf(s[0][v], s[1][v]);
            t = fmaxf(t, lanexor16<1>(t));
            t = fmaxf(t, lanexor16<2>(t));
            t = fmaxf(t, lanexor16<4>(t));
            t = fmaxf(t, lanexor16<8>(t));
            const float mn   = fmaxf(mrow[v], t);
            const float corr = __expf(mrow[v] - mn);   // exp(-inf)=0 first time
            mrow[v]   = mn;
            lrow[v]  *= corr;
            pscale[v] = corr;
        }
#pragma unroll
        for (int j = 0; j < 4; ++j)
#pragma unroll
            for (int v = 0; v < 8; ++v) accO[j][v] = accO[j][v] * pscale[v];

        // ---- P = exp(S - m), stage to wave-private LDS ----
#pragma unroll
        for (int jn = 0; jn < 2; ++jn)
#pragma unroll
            for (int v = 0; v < 8; ++v) {
                const float p = __expf(s[jn][v] - mrow[v]);
                Plds[wave][v + 8 * g2][jn * 16 + lm] = (f16)p;
            }
        v16h aP;
        {
            const f16* pp = &Plds[wave][lm][0];
            v8h lo = *(const v8h*)(pp + 16 * g2);
            v8h hi = *(const v8h*)(pp + 16 * g2 + 8);
#pragma unroll
            for (int i = 0; i < 8; ++i) { aP[i] = lo[i]; aP[i + 8] = hi[i]; }
        }

        // ---- row sums: P . ones via WMMA, lands in per-lane stat layout ----
        {
            v8f rsum = __builtin_amdgcn_wmma_f32_16x16x32_f16(
                false, aP, false, ones, (short)0, zero8, false, false);
#pragma unroll
            for (int v = 0; v < 8; ++v) lrow[v] += rsum[v];
        }

        // ---- O += P . V (V^T tile in LDS -> contiguous B-fragments) ----
#pragma unroll
        for (int j = 0; j < 4; ++j) {
            const f16* vp = &Vs[cur][j * 16 + lm][16 * g2];
            v8h lo = *(const v8h*)vp;
            v8h hi = *(const v8h*)(vp + 8);
            v16h bf;
#pragma unroll
            for (int i = 0; i < 8; ++i) { bf[i] = lo[i]; bf[i + 8] = hi[i]; }
            accO[j] = __builtin_amdgcn_wmma_f32_16x16x32_f16(
                false, aP, false, bf, (short)0, accO[j], false, false);
        }

        // next tile has landed (own transfers) + all waves done with cur
        wait_async0();
        __syncthreads();
        cur ^= 1;
    }

    // ---- epilogue: /l, query mask, write merged [B][T][512] f32 ----
    const float* qm = qmask + (size_t)b * TT;
    const int head  = hb >> 2;
#pragma unroll
    for (int v = 0; v < 8; ++v) {
        const int t   = q0 + v + 8 * g2;
        const float f = qm[t] / lrow[v];
#pragma unroll
        for (int j = 0; j < 4; ++j) {
            const int col = head * DH + j * 16 + lm;
            attnOut[(size_t)(b * TT + t) * DD + col] = accO[j][v] * f;
        }
    }
}

// ---------------------------------------------------------------------------
// Kernel 4: residual + LayerNorm (unbiased std, eps added to std).
// ---------------------------------------------------------------------------
__global__ __launch_bounds__(256)
void ln_kernel(const float* __restrict__ attn, const float* __restrict__ queries,
               const float* __restrict__ gamma, const float* __restrict__ beta,
               float* __restrict__ out) {
    const int row  = blockIdx.x * 8 + (threadIdx.x >> 5);
    const int lane = threadIdx.x & 31;
    const float* a = attn    + (size_t)row * DD;
    const float* q = queries + (size_t)row * DD;
    float x[16];
    float s = 0.f, s2 = 0.f;
#pragma unroll
    for (int kk = 0; kk < 4; ++kk) {
        const int c = 4 * (lane + 32 * kk);
        v4f va = *(const v4f*)(a + c);
        v4f vq = *(const v4f*)(q + c);
#pragma unroll
        for (int i = 0; i < 4; ++i) {
            const float t = va[i] + vq[i];
            x[kk * 4 + i] = t;
            s += t; s2 += t * t;
        }
    }
#pragma unroll
    for (int m = 16; m >= 1; m >>= 1) { s += __shfl_xor(s, m, 32); s2 += __shfl_xor(s2, m, 32); }
    const float mean = s / 512.f;
    float var = (s2 - 512.f * mean * mean) / 511.f;
    var = var < 0.f ? 0.f : var;
    const float inv = 1.f / (sqrtf(var) + 1e-8f);
#pragma unroll
    for (int kk = 0; kk < 4; ++kk) {
        const int c = 4 * (lane + 32 * kk);
#pragma unroll
        for (int i = 0; i < 4; ++i)
            out[(size_t)row * DD + c + i] =
                gamma[c + i] * (x[kk * 4 + i] - mean) * inv + beta[c + i];
    }
}

// ---------------------------------------------------------------------------
extern "C" void kernel_launch(void* const* d_in, const int* in_sizes, int n_in,
                              void* d_out, int out_size, void* d_ws, size_t ws_size,
                              hipStream_t stream) {
    const float* queries = (const float*)d_in[0];
    const float* keys    = (const float*)d_in[1];
    const float* values  = (const float*)d_in[2];
    const float* Wq      = (const float*)d_in[3];
    const float* bq      = (const float*)d_in[4];
    const float* Wk      = (const float*)d_in[5];
    const float* bk      = (const float*)d_in[6];
    const float* Wv      = (const float*)d_in[7];
    const float* bv      = (const float*)d_in[8];
    const float* gamma   = (const float*)d_in[9];
    const float* beta    = (const float*)d_in[10];

    char* ws = (char*)d_ws;
    f16*   Qh    = (f16*)(ws + 0);            //  8.39 MB  [32][2048][64] f16
    f16*   Kh    = (f16*)(ws + 8388608);      //  8.39 MB
    f16*   Vt    = (f16*)(ws + 16777216);     //  8.39 MB  [32][64][2048] f16
    float* attn  = (float*)(ws + 25165824);   // 16.78 MB  [4][2048][512] f32
    float* qmask = (float*)(ws + 41943040);   // 32 KB
    float* kmask = (float*)(ws + 41975808);   // 32 KB

    mask_kernel<<<2048, 256, 0, stream>>>(queries, keys, qmask, kmask);

    dim3 pgrid(8, 128);   // (N tiles, M tiles)
    proj_kernel<0><<<pgrid, 128, 0, stream>>>(queries, Wq, bq, Qh, 0.125f);
    proj_kernel<0><<<pgrid, 128, 0, stream>>>(keys,    Wk, bk, Kh, 1.0f);
    proj_kernel<1><<<pgrid, 128, 0, stream>>>(values,  Wv, bv, Vt, 1.0f);

    attn_kernel<<<1024, 128, 0, stream>>>(Qh, Kh, Vt, kmask, qmask, attn);

    ln_kernel<<<1024, 256, 0, stream>>>(attn, queries, gamma, beta, (float*)d_out);
}
